// WESUP_6038724018542
// MI455X (gfx1250) — matrix operand order repaired and checked by reading, not compiled, not run on hardware
//
#include <hip/hip_runtime.h>
#include <hip/hip_bf16.h>
#include <cstdint>
#include <cstddef>

// ---------------------------------------------------------------------------
// WESUP pipeline on gfx1250 (wave32, WMMA).
// All matmuls use v_wmma_f32_16x16x32_bf16 (bf16 A/B, f32 accumulate).
// Superpixel-pooling GEMM stages its bf16 B tiles with
// global_load_async_to_lds_b128 (ASYNCcnt) per CDNA5 ISA.
// ---------------------------------------------------------------------------

typedef __attribute__((ext_vector_type(16))) __bf16        v16bf;
typedef __attribute__((ext_vector_type(8)))  float         v8f;
typedef __attribute__((ext_vector_type(8)))  unsigned int  v8u;

#define HH    224
#define WW    224
#define HW0   (224 * 224)
#define NSP   1024
#define FSUM  2112

// fp32 -> bf16, round-to-nearest-even
__device__ __forceinline__ unsigned short f2bf(float f) {
  unsigned u = __float_as_uint(f);
  u += 0x7FFFu + ((u >> 16) & 1u);
  return (unsigned short)(u >> 16);
}

// A-matrix 16x32 bf16 fragment (ISA 7.12.2): lane half h, row M = lane&15.
// element e -> K = (e>>3)*16 + h*8 + (e&7). Pairs (even e) are contiguous -> u32 LDS reads.
__device__ __forceinline__ v16bf fragA(const unsigned short* lds, int row, int half) {
  const unsigned* p = (const unsigned*)(lds + row * 32);
  v8u t;
#pragma unroll
  for (int i = 0; i < 8; ++i) {
    int kk = (i >> 2) * 16 + half * 8 + (i & 3) * 2;
    t[i] = p[kk >> 1];
  }
  return __builtin_bit_cast(v16bf, t);
}

// B-matrix 32x16 bf16 fragment: column N = lane&15, K = (e>>3)*16 + half*8 + (e&7).
// LDS tile stored transposed: [n][k] row-major.
__device__ __forceinline__ v16bf fragB(const unsigned short* lds, int row, int half) {
  const unsigned* p = (const unsigned*)(lds + row * 32);
  v8u t;
#pragma unroll
  for (int i = 0; i < 8; ++i) {
    int kk = (i >> 2) * 16 + half * 8 + (i & 3) * 2;
    t[i] = p[kk >> 1];
  }
  return __builtin_bit_cast(v16bf, t);
}

// ---------------------------------------------------------------------------
// 3x3 conv (pad 1) as implicit GEMM: out(Cout,HW) = W(Cout,9*Cin) @ im2col + b
// One wave per 16(M) x 16(N-pixels) output tile. im2col built on the fly in LDS.
// ---------------------------------------------------------------------------
__global__ __launch_bounds__(32)
void k_conv3x3_wmma(const float* __restrict__ in, const float* __restrict__ w,
                    const float* __restrict__ bias, float* __restrict__ out,
                    int Cin, int Cout, int H, int W) {
  __shared__ unsigned short Als[16 * 32];
  __shared__ unsigned short Bls[16 * 32];
  const int lane = threadIdx.x;
  const int HWl  = H * W;
  const int p0   = blockIdx.x * 16;   // pixel tile
  const int m0   = blockIdx.y * 16;   // out-channel tile
  const int K    = Cin * 9;
  const int nChunks = (K + 31) >> 5;
  const int row = lane & 15, half = lane >> 4;
  v8f acc = {};
  for (int c = 0; c < nChunks; ++c) {
    const int k = (c << 5) + lane;
    const bool kin = (k < K);
    // A tile: weights, coalesced across lanes
#pragma unroll 4
    for (int m = 0; m < 16; ++m) {
      float v = kin ? w[(size_t)(m0 + m) * K + k] : 0.0f;
      Als[m * 32 + lane] = f2bf(v);
    }
    // B tile: im2col gather, stored transposed [pix][k]
    const int ci = k / 9, r9 = k - ci * 9;
    const int ky = r9 / 3 - 1, kx = r9 - (r9 / 3) * 3 - 1;
#pragma unroll 4
    for (int n = 0; n < 16; ++n) {
      const int p = p0 + n;
      float v = 0.0f;
      if (kin && p < HWl) {
        const int y = p / W + ky, x = p - (p / W) * W + kx;
        if (y >= 0 && y < H && x >= 0 && x < W)
          v = in[(size_t)ci * HWl + (size_t)y * W + x];
      }
      Bls[n * 32 + lane] = f2bf(v);
    }
    __syncthreads();
    v16bf a = fragA(Als, row, half);
    v16bf b = fragB(Bls, row, half);
    acc = __builtin_amdgcn_wmma_f32_16x16x32_bf16(false, a, false, b,
                                                  (short)0, acc, false, false);
    __syncthreads();
  }
  // C layout: lane l, reg r -> N = l&15, M = (l>>4)*8 + r
#pragma unroll
  for (int r = 0; r < 8; ++r) {
    const int m = m0 + half * 8 + r;
    const int p = p0 + row;
    if (p < HWl) out[(size_t)m * HWl + p] = acc[r] + bias[m];
  }
}

// ---------------------------------------------------------------------------
// Generic WMMA GEMM: C(M,N) = A(M,K) @ B + bias, optional ReLU.
//   B = fp32 row-major (K,N)          (side 1x1 convs, FC layers)
//   BIASMODE 0: none, 1: per-M, 2: per-N
// ---------------------------------------------------------------------------
template <int BIASMODE, bool RELU>
__global__ __launch_bounds__(32)
void k_gemm(const float* __restrict__ A, const float* __restrict__ B,
            const float* __restrict__ bias, float* __restrict__ C,
            int M, int N, int K) {
  __shared__ unsigned short Als[16 * 32];
  __shared__ unsigned short Bls[16 * 32];
  const int lane = threadIdx.x;
  const int n0 = blockIdx.x * 16, m0 = blockIdx.y * 16;
  const int row = lane & 15, half = lane >> 4;
  v8f acc = {};
  for (int k0 = 0; k0 < K; k0 += 32) {
    const int k = k0 + lane;
    const bool kin = (k < K);
#pragma unroll 4
    for (int m = 0; m < 16; ++m) {
      float v = kin ? A[(size_t)(m0 + m) * K + k] : 0.0f;
      Als[m * 32 + lane] = f2bf(v);
    }
#pragma unroll 4
    for (int n = 0; n < 16; ++n) {
      unsigned short bv = 0;
      if (kin && (n0 + n) < N)
        bv = f2bf(B[(size_t)k * N + (n0 + n)]);
      Bls[n * 32 + lane] = bv;
    }
    __syncthreads();
    v16bf a = fragA(Als, row, half);
    v16bf b = fragB(Bls, row, half);
    acc = __builtin_amdgcn_wmma_f32_16x16x32_bf16(false, a, false, b,
                                                  (short)0, acc, false, false);
    __syncthreads();
  }
#pragma unroll
  for (int r = 0; r < 8; ++r) {
    const int m = m0 + half * 8 + r;
    const int n = n0 + row;
    if (n < N) {
      float v = acc[r];
      if (BIASMODE == 1) v += bias[m];
      if (BIASMODE == 2) v += bias[n];
      if (RELU) v = v > 0.0f ? v : 0.0f;
      C[(size_t)m * N + n] = v;
    }
  }
}

// ---------------------------------------------------------------------------
// Superpixel-pooling GEMM: C(M,N) = A(M,K) @ B(N,K)^T
//   A = sp, fp32 (M,K) row-major;  B = feat, bf16 (N,K) row-major.
//   Exact tiling (M%16==0, N%64==0, K%32==0) -> no guards.
//   One wave computes a 16x64 tile: A fragment reused across 4 WMMAs.
//   B tiles (contiguous 64-byte K-slices per row) are staged memory->LDS with
//   global_load_async_to_lds_b128, tracked by ASYNCcnt (ISA 08_async_tensor).
// ---------------------------------------------------------------------------
__global__ __launch_bounds__(32)
void k_gemm_pool(const float* __restrict__ A, const unsigned short* __restrict__ B,
                 float* __restrict__ C, int M, int N, int K) {
  __shared__ alignas(16) unsigned short Als[16 * 32];
  __shared__ alignas(16) unsigned short Bls[64 * 32];
  const int lane = threadIdx.x;
  const int n0 = blockIdx.x * 64, m0 = blockIdx.y * 16;
  const int row = lane & 15, half = lane >> 4;
  const unsigned ldsB = (unsigned)(size_t)(void*)Bls;   // LDS byte offset of Bls
  v8f acc[4] = {};
  for (int k0 = 0; k0 < K; k0 += 32) {
    // ---- A tile 16x32: fp32 b128 loads -> packed bf16 b64 LDS stores ----
#pragma unroll
    for (int it = 0; it < 4; ++it) {
      const int f = it * 128 + lane * 4;          // flat fp32 element index
      const int m = f >> 5, kk = f & 31;
      float4 v = *(const float4*)(A + (size_t)(m0 + m) * K + k0 + kk);
      uint2 pk;
      pk.x = (unsigned)f2bf(v.x) | ((unsigned)f2bf(v.y) << 16);
      pk.y = (unsigned)f2bf(v.z) | ((unsigned)f2bf(v.w) << 16);
      *(uint2*)&Als[m * 32 + kk] = pk;
    }
    // prefetch next A chunk (global_prefetch_b8)
    __builtin_prefetch(A + (size_t)(m0 + (lane >> 1)) * K + k0 + 32, 0, 1);
    // ---- B tile 64 rows x 64 bytes: async copy straight into LDS ----
#pragma unroll
    for (int it = 0; it < 8; ++it) {
      const int o = it * 512 + lane * 16;         // byte offset within tile
      const int n = o >> 6, koff = o & 63;        // 64 B of K-slice per row
      unsigned long long ga =
          (unsigned long long)(size_t)(B + (size_t)(n0 + n) * K + k0) + (unsigned)koff;
      unsigned la = ldsB + (unsigned)o;
      asm volatile("global_load_async_to_lds_b128 %0, %1, off"
                   :: "v"(la), "v"(ga) : "memory");
    }
    asm volatile("s_wait_asynccnt 0x0" ::: "memory");
    __syncthreads();
    v16bf a = fragA(Als, row, half);
#pragma unroll
    for (int t = 0; t < 4; ++t) {
      v16bf b = fragB(Bls + t * 16 * 32, row, half);
      acc[t] = __builtin_amdgcn_wmma_f32_16x16x32_bf16(false, a, false, b,
                                                       (short)0, acc[t], false, false);
    }
    __syncthreads();
  }
#pragma unroll
  for (int t = 0; t < 4; ++t)
#pragma unroll
    for (int r = 0; r < 8; ++r)
      C[(size_t)(m0 + half * 8 + r) * N + (n0 + t * 16 + row)] = acc[t][r];
}

// ---------------------------------------------------------------------------
// Elementwise / pooling / resize / softmax / scatter kernels
// ---------------------------------------------------------------------------
__global__ void k_relu(const float* __restrict__ in, float* __restrict__ out, int n) {
  int i = blockIdx.x * blockDim.x + threadIdx.x;
  if (i < n) { float v = in[i]; out[i] = v > 0.0f ? v : 0.0f; }
}

__global__ void k_relu_pool(const float* __restrict__ in, float* __restrict__ out,
                            int C, int H, int W) {
  const int Ho = H >> 1, Wo = W >> 1;
  const int tot = C * Ho * Wo;
  int i = blockIdx.x * blockDim.x + threadIdx.x;
  if (i >= tot) return;
  const int c = i / (Ho * Wo), r = i - c * (Ho * Wo);
  const int y = r / Wo, x = r - y * Wo;
  const float* p = in + (size_t)c * H * W + (size_t)(2 * y) * W + 2 * x;
  float v = fmaxf(fmaxf(p[0], p[1]), fmaxf(p[W], p[W + 1]));
  out[i] = v > 0.0f ? v : 0.0f;
}

// bilinear (align_corners=True) upsample of side map into the bf16 feat matrix
__global__ void k_resize_feat(const float* __restrict__ side, unsigned short* __restrict__ feat,
                              int Cs, int h, int w, int chOff) {
  const int tot = Cs * HW0;
  int i = blockIdx.x * blockDim.x + threadIdx.x;
  if (i >= tot) return;
  const int c = i / HW0, r = i - c * HW0;
  const int oy = r / WW, ox = r - oy * WW;
  const float sy = (h > 1) ? (float)(h - 1) / (float)(HH - 1) : 0.0f;
  const float sx = (w > 1) ? (float)(w - 1) / (float)(WW - 1) : 0.0f;
  const float fy = oy * sy, fx = ox * sx;
  int y0 = (int)fy; int y1 = min(y0 + 1, h - 1);
  int x0 = (int)fx; int x1 = min(x0 + 1, w - 1);
  const float wy = fy - (float)y0, wx = fx - (float)x0;
  const float* p = side + (size_t)c * h * w;
  const float top = p[(size_t)y0 * w + x0] * (1.0f - wx) + p[(size_t)y0 * w + x1] * wx;
  const float bot = p[(size_t)y1 * w + x0] * (1.0f - wx) + p[(size_t)y1 * w + x1] * wx;
  feat[(size_t)(chOff + c) * HW0 + r] = f2bf(top * (1.0f - wy) + bot * wy);
}

// softmax over 2 classes, keep foreground probability
__global__ void k_softmax_p1(const float* __restrict__ logits, float* __restrict__ p1, int n) {
  int i = blockIdx.x * blockDim.x + threadIdx.x;
  if (i >= n) return;
  const float a = logits[2 * i], b = logits[2 * i + 1];
  const float m = fmaxf(a, b);
  const float ea = __expf(a - m), eb = __expf(b - m);
  p1[i] = eb / (ea + eb);
}

// pred = hard^T @ sp_pred[:,1]; exploit exactly-one-superpixel-per-pixel structure
__global__ void k_scatter(const float* __restrict__ sp, const float* __restrict__ p1,
                          float* __restrict__ out) {
  int pix = blockIdx.x * blockDim.x + threadIdx.x;
  if (pix >= HW0) return;
  float acc = 0.0f;
  for (int i = 0; i < NSP; ++i) {
    if (sp[(size_t)i * HW0 + pix] > 0.0f) acc += p1[i];
  }
  out[pix] = acc;
}

// ---------------------------------------------------------------------------
// Host orchestration
// ---------------------------------------------------------------------------
extern "C" void kernel_launch(void* const* d_in, const int* in_sizes, int n_in,
                              void* d_out, int out_size, void* d_ws, size_t ws_size,
                              hipStream_t stream) {
  (void)in_sizes; (void)n_in; (void)out_size; (void)ws_size;

  const float* x  = (const float*)d_in[0];
  const float* sp = (const float*)d_in[1];
  const float *convW[13], *convB[13], *sideW[13], *sideB[13], *fcW[4], *fcB[4];
  for (int i = 0; i < 13; ++i) {
    convW[i] = (const float*)d_in[2 + i];
    convB[i] = (const float*)d_in[15 + i];
    sideW[i] = (const float*)d_in[28 + i];
    sideB[i] = (const float*)d_in[41 + i];
  }
  for (int i = 0; i < 4; ++i) {
    fcW[i] = (const float*)d_in[54 + i];
    fcB[i] = (const float*)d_in[58 + i];
  }

  // carve workspace (every byte we read is written first; no init needed)
  char* ws = (char*)d_ws;
  auto carve = [&](size_t bytes) {
    void* p = (void*)ws;
    ws += (bytes + 255) & ~(size_t)255;
    return p;
  };
  unsigned short* feat = (unsigned short*)carve((size_t)FSUM * HW0 * 2); // 202 MiB, bf16
  float* actA  = (float*)carve((size_t)64 * HW0 * 4);
  float* actB  = (float*)carve((size_t)64 * HW0 * 4);
  float* convO = (float*)carve((size_t)64 * HW0 * 4);
  float* sideO = (float*)carve((size_t)32 * HW0 * 4);
  float* fbuf0 = (float*)carve((size_t)NSP * FSUM * 4);
  float* fbuf1 = (float*)carve((size_t)NSP * FSUM * 4);
  float* p1buf = (float*)carve((size_t)NSP * 4);

  static const int  cfg[13] = {64,64,128,128,256,256,256,512,512,512,512,512,512};
  static const int  hs[13]  = {224,224,112,112,56,56,56,28,28,28,14,14,14};
  static const bool pl[13]  = {false,true,false,true,false,false,true,false,false,true,false,false,true};

  const float* cur = x;
  int Cin = 3, chOff = 0;
  for (int i = 0; i < 13; ++i) {
    const int Cout = cfg[i], H = hs[i], W = hs[i], HWl = H * W;
    // conv (pre-ReLU output kept)
    dim3 gConv((HWl + 15) / 16, Cout / 16);
    k_conv3x3_wmma<<<gConv, 32, 0, stream>>>(cur, convW[i], convB[i], convO, Cin, Cout, H, W);
    // 1x1 side conv: (Cout/2, HWl) = side_w(Cout/2, Cout) @ convO(Cout, HWl)
    const int Ms = Cout / 2;
    dim3 gSide((HWl + 15) / 16, Ms / 16);
    k_gemm<1, false><<<gSide, 32, 0, stream>>>(sideW[i], convO, sideB[i], sideO, Ms, HWl, Cout);
    // bilinear upsample into bf16 feat rows [chOff, chOff+Ms)
    const int totR = Ms * HW0;
    k_resize_feat<<<(totR + 255) / 256, 256, 0, stream>>>(sideO, feat, Ms, H, W, chOff);
    chOff += Ms;
    // ReLU (+ optional 2x2 maxpool) to produce next-layer input
    float* nxt = (cur == actA) ? actB : actA;
    if (pl[i]) {
      const int tot = Cout * (H / 2) * (W / 2);
      k_relu_pool<<<(tot + 255) / 256, 256, 0, stream>>>(convO, nxt, Cout, H, W);
    } else {
      const int tot = Cout * HWl;
      k_relu<<<(tot + 255) / 256, 256, 0, stream>>>(convO, nxt, tot);
    }
    cur = nxt;
    Cin = Cout;
  }

  // superpixel mean pooling: f(1024,2112) = sp(1024,50176) @ feat^T
  // (16x64 tiles, async-LDS B staging; N=2112=33*64, K=50176=1568*32, exact)
  dim3 gPool(FSUM / 64, NSP / 16);
  k_gemm_pool<<<gPool, 32, 0, stream>>>(sp, feat, fbuf0, NSP, FSUM, HW0);

  // MLP head (bias per output column, ReLU on first three)
  k_gemm<2, true ><<<dim3(1024 / 16, NSP / 16), 32, 0, stream>>>(fbuf0, fcW[0], fcB[0], fbuf1, NSP, 1024, FSUM);
  k_gemm<2, true ><<<dim3(1024 / 16, NSP / 16), 32, 0, stream>>>(fbuf1, fcW[1], fcB[1], fbuf0, NSP, 1024, 1024);
  k_gemm<2, true ><<<dim3(  32 / 16, NSP / 16), 32, 0, stream>>>(fbuf0, fcW[2], fcB[2], fbuf1, NSP,   32, 1024);
  k_gemm<2, false><<<dim3(        1, NSP / 16), 32, 0, stream>>>(fbuf1, fcW[3], fcB[3], fbuf0, NSP,    2,   32);

  k_softmax_p1<<<(NSP + 255) / 256, 256, 0, stream>>>(fbuf0, p1buf, NSP);
  k_scatter<<<(HW0 + 255) / 256, 256, 0, stream>>>(sp, p1buf, (float*)d_out);
}
